// Qwen3EncoderAttention_64218351009812
// MI455X (gfx1250) — matrix operand order repaired and checked by reading, not compiled, not run on hardware
//
#include <hip/hip_runtime.h>

typedef __attribute__((ext_vector_type(16))) __bf16 v16bf;
typedef __attribute__((ext_vector_type(8)))  float  v8f;
typedef unsigned short u16;

union Frag { v16bf v; uint4 q[2]; };

enum : int {
  B_ = 2, S_ = 2048, HID_ = 2048, NH_ = 16, NKV_ = 8, D_ = 128,
  BS_ = B_ * S_
};

__device__ __forceinline__ u16 f2bf(float x) {
  unsigned int u = __float_as_uint(x);
  unsigned int r = u + 0x7FFFu + ((u >> 16) & 1u);   // round-to-nearest-even
  return (u16)(r >> 16);
}

__device__ __forceinline__ v8f wmma_bf16(Frag a, Frag b, v8f c) {
  return __builtin_amdgcn_wmma_f32_16x16x32_bf16(false, a.v, false, b.v,
                                                 (short)0, c, false, false);
}

// CDNA5 async copy global -> LDS, 16B per lane, tracked by ASYNCcnt.
// LDS byte offset == low 32 bits of the flat shared-memory address (ISA 10.2).
__device__ __forceinline__ void async_ld_b128(void* lds_ptr, const void* gptr) {
  unsigned int loff = (unsigned int)(uintptr_t)lds_ptr;
  unsigned long long ga = (unsigned long long)(uintptr_t)gptr;
  asm volatile("global_load_async_to_lds_b128 %0, %1, off"
               :: "v"(loff), "v"(ga) : "memory");
}
__device__ __forceinline__ void wait_async0() {
  asm volatile("s_wait_asynccnt 0x0" ::: "memory");
}

// ---------------- elementwise fp32 -> bf16 ----------------
__global__ void k_tobf16(const float* __restrict__ src, u16* __restrict__ dst, int n) {
  int i = blockIdx.x * 256 + threadIdx.x;
  if (i < n) dst[i] = f2bf(src[i]);
}

// ---------------- transpose KxN fp32 -> NxK bf16 ----------------
__global__ void k_transpose_bf16(const float* __restrict__ src, u16* __restrict__ dst,
                                 int K, int N) {
  int idx = blockIdx.x * 256 + threadIdx.x;
  if (idx >= K * N) return;
  int n = idx % N, k = idx / N;
  dst[(size_t)n * K + k] = f2bf(src[(size_t)k * N + n]);
}

// ---------------- generic WMMA GEMM: C(MxN,f32) = A(MxK,bf16) * Bt(NxK,bf16)^T ----
// 128x128x64 tiles, 8 waves, double-buffered async-to-LDS staging.
__global__ __launch_bounds__(256) void k_gemm_bf16(const u16* __restrict__ A,
                                                   const u16* __restrict__ Bt,
                                                   float* __restrict__ C,
                                                   int M, int N, int K) {
  __shared__ u16 Al[2][128 * 72];     // [row][k], pad 8
  __shared__ u16 Bl[2][128 * 72];     // [col][k], pad 8
  const int tid  = threadIdx.x;
  const int wave = tid >> 5;
  const int lane = tid & 31;
  const int lrow = lane & 15, lhi = lane >> 4;
  const int m0 = blockIdx.y * 128;
  const int n0 = blockIdx.x * 128;

  auto stage = [&](int k0, int buf) {
#pragma unroll
    for (int i = 0; i < 4; ++i) {                 // 1024 chunks of 8 halves per tile
      int chunk = tid + i * 256;
      int row = chunk >> 3, seg = chunk & 7;
      async_ld_b128(&Al[buf][row * 72 + seg * 8],
                    A + (size_t)(m0 + row) * K + k0 + seg * 8);
      async_ld_b128(&Bl[buf][row * 72 + seg * 8],
                    Bt + (size_t)(n0 + row) * K + k0 + seg * 8);
    }
  };

  v8f acc[8];
#pragma unroll
  for (int t = 0; t < 8; ++t) acc[t] = (v8f){0, 0, 0, 0, 0, 0, 0, 0};

  stage(0, 0);
  int cur = 0;
  for (int k0 = 0; k0 < K; k0 += 64) {
    wait_async0();
    __syncthreads();
    if (k0 + 64 < K) stage(k0 + 64, cur ^ 1);

    const int arow = wave * 16 + lrow;
    const int kb = lhi * 8;
#pragma unroll
    for (int c = 0; c < 2; ++c) {
      Frag a;
      a.q[0] = *reinterpret_cast<const uint4*>(&Al[cur][arow * 72 + c * 32 + kb]);
      a.q[1] = *reinterpret_cast<const uint4*>(&Al[cur][arow * 72 + c * 32 + 16 + kb]);
      Frag bq[8];
#pragma unroll
      for (int nt = 0; nt < 8; ++nt) {
        const int brow = nt * 16 + lrow;
        bq[nt].q[0] =
            *reinterpret_cast<const uint4*>(&Bl[cur][brow * 72 + c * 32 + lhi * 8]);
        bq[nt].q[1] =
            *reinterpret_cast<const uint4*>(&Bl[cur][brow * 72 + c * 32 + 16 + lhi * 8]);
      }
#pragma unroll
      for (int nt = 0; nt < 8; ++nt) acc[nt] = wmma_bf16(a, bq[nt], acc[nt]);
    }
    cur ^= 1;
  }

#pragma unroll
  for (int nt = 0; nt < 8; ++nt)
#pragma unroll
    for (int j = 0; j < 8; ++j) {
      int row = m0 + wave * 16 + j + lhi * 8;
      C[(size_t)row * N + n0 + nt * 16 + lrow] = acc[nt][j];
    }
}

// ---------------- per-head RMSNorm + RoPE, fp32 in -> bf16 [b][h][s][d] -----------
__global__ __launch_bounds__(128) void k_norm_rope(const float* __restrict__ src,
                                                   const float* __restrict__ w,
                                                   u16* __restrict__ dst, int nh) {
  const int tid = threadIdx.x;
  const int idx = blockIdx.x;
  const int h = idx % nh;
  const int bs = idx / nh;
  const int s = bs % S_;
  const int b = bs / S_;

  float x = src[(size_t)bs * (nh * D_) + h * D_ + tid];
  float ss = x * x;
#pragma unroll
  for (int off = 16; off >= 1; off >>= 1) ss += __shfl_xor(ss, off, 32);

  __shared__ float red[4];
  __shared__ float xs[128];
  if ((tid & 31) == 0) red[tid >> 5] = ss;
  __syncthreads();
  float tot = red[0] + red[1] + red[2] + red[3];
  float xn = x * rsqrtf(tot * (1.0f / 128.0f) + 1e-6f) * w[tid];
  xs[tid] = xn;
  __syncthreads();
  float other = xs[tid ^ 64];

  int j = tid & 63;
  float inv = __expf(-0.1439115681f * (float)j);  // theta^(-2j/128)
  float ang = (float)s * inv;
  float c = cosf(ang), sn = sinf(ang);
  float out = (tid < 64) ? (xn * c - other * sn) : (xn * c + other * sn);
  dst[(((size_t)(b * nh + h)) * S_ + s) * D_ + tid] = f2bf(out);
}

// ---------------- V: fp32 [b][s][kvh][d] -> bf16 [b][kvh][d][s] -------------------
__global__ void k_vpack(const float* __restrict__ V, u16* __restrict__ Vt) {
  int idx = blockIdx.x * 256 + threadIdx.x;     // B*S*NKV*D threads
  int d = idx & (D_ - 1);
  int r = idx >> 7;
  int kvh = r & (NKV_ - 1); r >>= 3;
  int s = r & (S_ - 1);
  int b = r >> 11;
  float v = V[(((size_t)(b * S_ + s)) * NKV_ + kvh) * D_ + d];
  Vt[(((size_t)(b * NKV_ + kvh)) * D_ + d) * S_ + s] = f2bf(v);
}

// ---------------- flash attention (bidirectional, additive pad mask) --------------
// Q: bf16 [b][h][s][d]; K: bf16 [b][kvh][s][d]; Vt: bf16 [b][kvh][d][s]
// O: bf16 [b][s][h*D+d]   (ready as A-matrix for the output projection)
// 64 queries/WG (16/wave), 128-key blocks, double-buffered async staging.
__global__ __launch_bounds__(128) void k_attn(const u16* __restrict__ Q,
                                              const u16* __restrict__ Kb,
                                              const u16* __restrict__ Vt,
                                              const float* __restrict__ amask,
                                              u16* __restrict__ O) {
  __shared__ u16 Kl[2][128 * 136];    // [key][d], pad 8
  __shared__ u16 Vl[2][128 * 136];    // [d][key], pad 8
  __shared__ u16 Pl[4 * 16 * 136];    // per-wave P scratch [q][key]

  const int tid = threadIdx.x, wave = tid >> 5, lane = tid & 31;
  const int lrow = lane & 15, lhi = lane >> 4;
  const int qb0 = blockIdx.x * 64;
  const int h = blockIdx.y, b = blockIdx.z;
  const int kvh = h >> 1;             // N_REP = 2
  const float SCALE = 0.08838834764831845f;   // 1/sqrt(128)

  const u16* Kbase = Kb + (size_t)(b * NKV_ + kvh) * S_ * D_;
  const u16* Vbase = Vt + (size_t)(b * NKV_ + kvh) * D_ * S_;
  u16* Pw = &Pl[wave * 16 * 136];

  auto stage = [&](int kb0, int buf) {
#pragma unroll
    for (int i = 0; i < 16; ++i) {                    // 2048 chunks per tile
      int chunk = tid + i * 128;
      int row = chunk >> 4, seg = chunk & 15;         // 16 chunks per 128-half row
      async_ld_b128(&Kl[buf][row * 136 + seg * 8],
                    Kbase + (size_t)(kb0 + row) * D_ + seg * 8);
      async_ld_b128(&Vl[buf][row * 136 + seg * 8],
                    Vbase + (size_t)row * S_ + kb0 + seg * 8);
    }
  };

  // this wave's Q A-fragments (16 rows x 128 d) straight from global
  const u16* Qbase = Q + ((size_t)(b * NH_ + h) * S_ + (qb0 + wave * 16 + lrow)) * D_;
  Frag qf[4];
#pragma unroll
  for (int c = 0; c < 4; ++c) {
    qf[c].q[0] = *reinterpret_cast<const uint4*>(Qbase + c * 32 + lhi * 8);
    qf[c].q[1] = *reinterpret_cast<const uint4*>(Qbase + c * 32 + 16 + lhi * 8);
  }

  v8f o[8];
  float m[8], l[8];
#pragma unroll
  for (int t = 0; t < 8; ++t) o[t] = (v8f){0, 0, 0, 0, 0, 0, 0, 0};
#pragma unroll
  for (int j = 0; j < 8; ++j) { m[j] = -1e30f; l[j] = 0.0f; }

  stage(0, 0);
  int cur = 0;
  for (int kb0 = 0; kb0 < S_; kb0 += 128) {
    wait_async0();
    __syncthreads();
    if (kb0 + 128 < S_) stage(kb0 + 128, cur ^ 1);

    // scores: 16 q x 128 k
    v8f st[8];
#pragma unroll
    for (int kt = 0; kt < 8; ++kt) st[kt] = (v8f){0, 0, 0, 0, 0, 0, 0, 0};
#pragma unroll
    for (int c = 0; c < 4; ++c) {
#pragma unroll
      for (int kt = 0; kt < 8; ++kt) {
        Frag kf;
        const int krow = kt * 16 + lrow;
        kf.q[0] =
            *reinterpret_cast<const uint4*>(&Kl[cur][krow * 136 + c * 32 + lhi * 8]);
        kf.q[1] =
            *reinterpret_cast<const uint4*>(&Kl[cur][krow * 136 + c * 32 + 16 + lhi * 8]);
        st[kt] = wmma_bf16(qf[c], kf, st[kt]);
      }
    }

    // scale + additive padding mask
    float addm[8];
#pragma unroll
    for (int kt = 0; kt < 8; ++kt)
      addm[kt] = (1.0f - amask[b * S_ + kb0 + kt * 16 + lrow]) * -1e30f;
#pragma unroll
    for (int kt = 0; kt < 8; ++kt)
#pragma unroll
      for (int j = 0; j < 8; ++j) st[kt][j] = st[kt][j] * SCALE + addm[kt];

    // online softmax: rows live in 16-lane halves of the C layout
    float mn[8], corr[8], bs[8];
#pragma unroll
    for (int j = 0; j < 8; ++j) {
      float v = st[0][j];
#pragma unroll
      for (int kt = 1; kt < 8; ++kt) v = fmaxf(v, st[kt][j]);
#pragma unroll
      for (int off = 8; off >= 1; off >>= 1) v = fmaxf(v, __shfl_xor(v, off, 16));
      mn[j] = fmaxf(m[j], v);
      corr[j] = __expf(m[j] - mn[j]);
      m[j] = mn[j];
      bs[j] = 0.0f;
    }
#pragma unroll
    for (int kt = 0; kt < 8; ++kt)
#pragma unroll
      for (int j = 0; j < 8; ++j) {
        float p = __expf(st[kt][j] - mn[j]);
        bs[j] += p;
        Pw[(j + lhi * 8) * 136 + kt * 16 + lrow] = f2bf(p);   // C-layout -> LDS
      }
#pragma unroll
    for (int j = 0; j < 8; ++j) {
      float v = bs[j];
#pragma unroll
      for (int off = 8; off >= 1; off >>= 1) v += __shfl_xor(v, off, 16);
      l[j] = l[j] * corr[j] + v;
    }
#pragma unroll
    for (int dt = 0; dt < 8; ++dt)
#pragma unroll
      for (int j = 0; j < 8; ++j) o[dt][j] *= corr[j];

    // P(16x128) @ V(128x128)
#pragma unroll
    for (int c = 0; c < 4; ++c) {
      Frag pa;
      pa.q[0] = *reinterpret_cast<const uint4*>(&Pw[lrow * 136 + c * 32 + lhi * 8]);
      pa.q[1] = *reinterpret_cast<const uint4*>(&Pw[lrow * 136 + c * 32 + lhi * 8 + 16]);
#pragma unroll
      for (int dt = 0; dt < 8; ++dt) {
        Frag vb;
        const int vrow = dt * 16 + lrow;
        vb.q[0] =
            *reinterpret_cast<const uint4*>(&Vl[cur][vrow * 136 + c * 32 + lhi * 8]);
        vb.q[1] =
            *reinterpret_cast<const uint4*>(&Vl[cur][vrow * 136 + c * 32 + 16 + lhi * 8]);
        o[dt] = wmma_bf16(pa, vb, o[dt]);
      }
    }
    cur ^= 1;
  }

  // epilogue: normalize and store bf16 [b][s][h*D+d]
#pragma unroll
  for (int j = 0; j < 8; ++j) l[j] = 1.0f / l[j];
#pragma unroll
  for (int dt = 0; dt < 8; ++dt)
#pragma unroll
    for (int j = 0; j < 8; ++j) {
      int srow = qb0 + wave * 16 + j + lhi * 8;
      O[((size_t)(b * S_ + srow)) * (NH_ * D_) + h * D_ + dt * 16 + lrow] =
          f2bf(o[dt][j] * l[j]);
    }
}

// =================================================================================
extern "C" void kernel_launch(void* const* d_in, const int* in_sizes, int n_in,
                              void* d_out, int out_size, void* d_ws, size_t ws_size,
                              hipStream_t stream) {
  (void)in_sizes; (void)n_in; (void)out_size; (void)ws_size;
  const float* hidden = (const float*)d_in[0];
  const float* amask  = (const float*)d_in[1];
  const float* wq     = (const float*)d_in[2];
  const float* wk     = (const float*)d_in[3];
  const float* wv     = (const float*)d_in[4];
  const float* wo     = (const float*)d_in[5];
  const float* qnw    = (const float*)d_in[6];
  const float* knw    = (const float*)d_in[7];

  char* ws = (char*)d_ws;
  size_t off = 0;
  auto take = [&](size_t bytes) -> char* {
    char* p = ws + off;
    off += (bytes + 255) & ~(size_t)255;
    return p;
  };

  u16*   Xb  = (u16*)take((size_t)BS_ * HID_ * 2);          // hidden bf16
  u16*   WqT = (u16*)take((size_t)2048 * 2048 * 2);
  u16*   WkT = (u16*)take((size_t)1024 * 2048 * 2);
  u16*   WvT = (u16*)take((size_t)1024 * 2048 * 2);
  u16*   WoT = (u16*)take((size_t)2048 * 2048 * 2);
  float* Qf  = (float*)take((size_t)BS_ * 2048 * 4);
  float* Kf  = (float*)take((size_t)BS_ * 1024 * 4);
  float* Vf  = (float*)take((size_t)BS_ * 1024 * 4);
  u16*   Qb  = (u16*)take((size_t)BS_ * 2048 * 2);          // [b][h][s][d]
  u16*   Kb  = (u16*)take((size_t)BS_ * 1024 * 2);          // [b][kvh][s][d]
  u16*   Vt  = (u16*)take((size_t)BS_ * 1024 * 2);          // [b][kvh][d][s]
  u16*   Ob  = (u16*)take((size_t)BS_ * 2048 * 2);          // attn out bf16

  // 1. casts / weight transposes
  k_tobf16<<<(BS_ * HID_) / 256, 256, 0, stream>>>(hidden, Xb, BS_ * HID_);
  k_transpose_bf16<<<(2048 * 2048) / 256, 256, 0, stream>>>(wq, WqT, 2048, 2048);
  k_transpose_bf16<<<(2048 * 1024) / 256, 256, 0, stream>>>(wk, WkT, 2048, 1024);
  k_transpose_bf16<<<(2048 * 1024) / 256, 256, 0, stream>>>(wv, WvT, 2048, 1024);
  k_transpose_bf16<<<(2048 * 2048) / 256, 256, 0, stream>>>(wo, WoT, 2048, 2048);

  // 2. QKV projections (WMMA)
  k_gemm_bf16<<<dim3(16, 32), 256, 0, stream>>>(Xb, WqT, Qf, BS_, 2048, 2048);
  k_gemm_bf16<<<dim3(8, 32), 256, 0, stream>>>(Xb, WkT, Kf, BS_, 1024, 2048);
  k_gemm_bf16<<<dim3(8, 32), 256, 0, stream>>>(Xb, WvT, Vf, BS_, 1024, 2048);

  // 3. QK RMSNorm + RoPE; V repack
  k_norm_rope<<<BS_ * NH_, 128, 0, stream>>>(Qf, qnw, Qb, NH_);
  k_norm_rope<<<BS_ * NKV_, 128, 0, stream>>>(Kf, knw, Kb, NKV_);
  k_vpack<<<(BS_ * NKV_ * D_) / 256, 256, 0, stream>>>(Vf, Vt);

  // 4. flash attention (WMMA QK^T + online softmax + WMMA PV)
  k_attn<<<dim3(S_ / 64, NH_, B_), 128, 0, stream>>>(Qb, Kb, Vt, amask, Ob);

  // 5. output projection (WMMA) -> fp32 d_out
  k_gemm_bf16<<<dim3(16, 32), 256, 0, stream>>>(Ob, WoT, (float*)d_out, BS_, 2048, 2048);
}
